// FourLayer_64F_22694607192114
// MI455X (gfx1250) — matrix operand order repaired and checked by reading, not compiled, not run on hardware
//
#include <hip/hip_runtime.h>

typedef __attribute__((ext_vector_type(16))) _Float16 v16h;
typedef __attribute__((ext_vector_type(8)))  _Float16 v8h;
typedef __attribute__((ext_vector_type(8)))  float    v8f;

#define NIMG   57          // 32 query + 25 support images
#define NQIMG  32
#define NCLS   5
#define SHOT   5
#define CH     64
#define H0     84
#define H1     42
#define H2     21
#define HW0    (H0*H0)     // 7056
#define HW1    (H1*H1)     // 1764
#define HW2    (H2*H2)     // 441
#define M_SUP  (SHOT*HW2)  // 2205
#define NQT    28          // ceil(441/16)
#define NMT    138         // ceil(2205/16)
#define NT1    111         // ceil(1764/16)
#define NT2    28          // ceil(441/16)
#define WPK_ELEMS (4*9*2*32*16)   // packed weight halfs per layer = 36864
#define WPK_BYTES (WPK_ELEMS*2)   // 73728 B staged in LDS per workgroup
#define NEG_BIG  (-3.4e38f)

__device__ __forceinline__ int bn_group(int img){ return img < NQIMG ? 0 : 1 + (img - NQIMG)/SHOT; }

__device__ __forceinline__ v16h zero16(){
  v16h z;
  #pragma unroll
  for (int i = 0; i < 16; ++i) z[i] = (_Float16)0.f;
  return z;
}

__device__ __forceinline__ v8f wmma_f16(v16h a, v16h b, v8f c){
  // D = A(16x32 f16) * B(32x16 f16) + C(16x16 f32)
  return __builtin_amdgcn_wmma_f32_16x16x32_f16(false, a, false, b, (short)0, c, false, false);
}

// branchless sorted-triple insert (t0 >= t1 >= t2): pure v_min/v_max dataflow,
// co-executes with the WMMA (XDL) pipe instead of serializing on EXEC updates.
__device__ __forceinline__ void top3_insert(float v, float& t0, float& t1, float& t2){
  float m0 = fminf(t0, v);  t0 = fmaxf(t0, v);
  float m1 = fminf(t1, m0); t1 = fmaxf(t1, m0);
  t2 = fmaxf(t2, m1);
}

// ---------------------------------------------------------------- init
__global__ void zero_init(float* stats, float* out){
  int i = blockIdx.x * 256 + threadIdx.x;
  if (i < 4*6*CH*2) stats[i] = 0.f;
  if (i < NQIMG*NCLS) out[i] = 0.f;
}

// ------------------------------------------------- weight packing (A-fragment order)
// wp index = (((otile*9+tap)*2+kstep)*32+lane)*16 + e ; 16 halfs per lane = one v16h
__global__ void pack_w(const float* __restrict__ w, _Float16* __restrict__ wp){
  int idx = blockIdx.x * 256 + threadIdx.x;
  if (idx >= WPK_ELEMS) return;
  int e    = idx & 15;
  int t    = idx >> 4;
  int lane = t & 31;  t >>= 5;
  int ks   = t & 1;   t >>= 1;
  int tap  = t % 9;
  int ot   = t / 9;
  int m = lane & 15;
  int K;                       // contiguous K order matching B-frag
  if (lane < 16) K = e;          // K 0..15
  else           K = e + 16;     // K 16..31
  int ci = ks * 32 + K;
  int o  = ot * 16 + m;
  int ky = tap / 3, kx = tap % 3;
  wp[idx] = (_Float16)w[((o*CH + ci)*3 + ky)*3 + kx];
}

// ---------------------------------------------------------------- layer1 (direct conv)
__device__ __forceinline__ float conv1_at(const float* __restrict__ src,
                                          const float* __restrict__ w, int o, int y, int x){
  float acc = 0.f;
  #pragma unroll
  for (int c = 0; c < 3; ++c)
    #pragma unroll
    for (int ky = 0; ky < 3; ++ky){
      int ys = y + ky - 1; if (ys < 0 || ys >= H0) continue;
      #pragma unroll
      for (int kx = 0; kx < 3; ++kx){
        int xs = x + kx - 1; if (xs < 0 || xs >= H0) continue;
        acc = fmaf(w[((o*3 + c)*3 + ky)*3 + kx], src[(c*H0 + ys)*H0 + xs], acc);
      }
    }
  return acc;
}

__global__ void conv1_stats(const float* __restrict__ in1, const float* __restrict__ in2,
                            const float* __restrict__ w1, float* __restrict__ stats){
  __shared__ float ls[CH], lq[CH];
  int tid = threadIdx.x;
  if (tid < CH){ ls[tid] = 0.f; lq[tid] = 0.f; }
  __syncthreads();
  int img = blockIdx.y;
  const float* src = (img < NQIMG) ? in1 + (size_t)img*3*HW0
                                   : in2 + (size_t)(img - NQIMG)*3*HW0;
  int o = tid & 63, sub = tid >> 6;
  float s = 0.f, q = 0.f;
  for (int i = 0; i < 16; ++i){
    int p = blockIdx.x*64 + sub*16 + i;
    if (p < HW0){
      float v = conv1_at(src, w1, o, p / H0, p % H0);
      s += v; q += v*v;
    }
  }
  atomicAdd(&ls[o], s); atomicAdd(&lq[o], q);
  __syncthreads();
  if (tid < CH){
    int g = bn_group(img);
    atomicAdd(&stats[(g*CH + tid)*2 + 0], ls[tid]);
    atomicAdd(&stats[(g*CH + tid)*2 + 1], lq[tid]);
  }
}

__global__ void bn_finalize(const float* __restrict__ stats, float* __restrict__ ss,
                            const float* __restrict__ gamma, const float* __restrict__ beta,
                            int HW){
  int i = blockIdx.x*64 + threadIdx.x;     // <<<6,64>>> : 6 groups x 64 ch
  int ch = i % CH;
  float cnt  = ((i / CH) == 0 ? (float)NQIMG : (float)SHOT) * (float)HW;
  float mean = stats[i*2 + 0] / cnt;
  float var  = stats[i*2 + 1] / cnt - mean*mean;
  float rstd = rsqrtf(var + 1e-5f);
  float sc   = gamma[ch] * rstd;
  ss[i*2 + 0] = sc;
  ss[i*2 + 1] = beta[ch] - mean*sc;
}

// recompute conv1, apply BN + LeakyReLU + 2x2 maxpool -> f16 NHWC act1 [57][42][42][64]
__global__ void apply1(const float* __restrict__ in1, const float* __restrict__ in2,
                       const float* __restrict__ w1, const float* __restrict__ ss,
                       _Float16* __restrict__ act1){
  int idx = blockIdx.x*256 + threadIdx.x;
  if (idx >= NIMG*HW1*CH) return;
  int o = idx & 63; int t = idx >> 6;
  int x2 = t % H1; t /= H1;
  int y2 = t % H1; int img = t / H1;
  const float* src = (img < NQIMG) ? in1 + (size_t)img*3*HW0
                                   : in2 + (size_t)(img - NQIMG)*3*HW0;
  int g = bn_group(img);
  float sc = ss[(g*CH + o)*2 + 0], sh = ss[(g*CH + o)*2 + 1];
  float m = NEG_BIG;
  #pragma unroll
  for (int py = 0; py < 2; ++py)
    #pragma unroll
    for (int px = 0; px < 2; ++px){
      float v = conv1_at(src, w1, o, 2*y2 + py, 2*x2 + px) * sc + sh;
      v = (v >= 0.f) ? v : 0.2f * v;
      m = fmaxf(m, v);
    }
  act1[idx] = (_Float16)m;
}

// ------------------------------------------- WMMA implicit-GEMM conv 64->64, 3x3 SAME
// one wave per 16-pixel tile (blockIdx.y = image); packed weights staged in LDS once
// per workgroup (8 waves share); D[o(64), p(16)] accumulated over 9 taps x 2 k-steps.
template<int H, int NT>
__global__ void __launch_bounds__(256) conv_wmma(const _Float16* __restrict__ act,
                                                 const _Float16* __restrict__ wp,
                                                 float* __restrict__ outbuf){
  constexpr int W  = H;
  constexpr int HW = H*W;
  extern __shared__ uint4 lwp4[];                 // 73728 B: full packed weight array
  const uint4* gwp4 = (const uint4*)wp;
  #pragma unroll 2
  for (int i = threadIdx.x; i < WPK_ELEMS/8; i += 256) lwp4[i] = gwp4[i];
  __syncthreads();
  const v16h* lwpv = (const v16h*)lwp4;

  int lane = threadIdx.x & 31;
  int tile = blockIdx.x*8 + (threadIdx.x >> 5);
  if (tile >= NT) return;                          // uniform per wave
  int img  = blockIdx.y;
  int p = tile*16 + (lane & 15);
  bool pvalid = p < HW;
  int y = p / W, x = p % W;                        // constant divisor -> mul/shift
  const _Float16* ibase = act + (size_t)img*HW*CH;
  int ciofB = (lane & 16) ? 16 : 0;                // B-frag ci sub-range per lane half
  v8f acc[4] = {};
  #pragma unroll
  for (int tap = 0; tap < 9; ++tap){
    int ys = y + tap/3 - 1, xs = x + tap%3 - 1;
    bool vld = pvalid && ys >= 0 && ys < H && xs >= 0 && xs < W;
    #pragma unroll
    for (int ks = 0; ks < 2; ++ks){
      v16h bfrag;
      if (vld) bfrag = *(const v16h*)(ibase + (size_t)(ys*W + xs)*CH + ks*32 + ciofB);
      else     bfrag = zero16();
      #pragma unroll
      for (int ot = 0; ot < 4; ++ot){
        v16h afrag = lwpv[((ot*9 + tap)*2 + ks)*32 + lane];   // ds_load from LDS
        acc[ot] = wmma_f16(afrag, bfrag, acc[ot]);
      }
    }
  }
  if (pvalid){
    float* ob = outbuf + ((size_t)img*HW + p)*CH + ((lane & 16) ? 8 : 0);
    #pragma unroll
    for (int ot = 0; ot < 4; ++ot) *(v8f*)(ob + ot*16) = acc[ot];
  }
}

// ---------------------------------------- BN stats over stored f32 conv output (NHWC)
__global__ void stats_buf(const float* __restrict__ buf, float* __restrict__ stats, int HW){
  __shared__ float ls[CH], lq[CH];
  int tid = threadIdx.x;
  if (tid < CH){ ls[tid] = 0.f; lq[tid] = 0.f; }
  __syncthreads();
  int img = blockIdx.y;
  int ch = tid & 63, sub = tid >> 6;
  const float* base = buf + (size_t)img*HW*CH;
  float s = 0.f, q = 0.f;
  #pragma unroll
  for (int j = 0; j < 16; ++j){
    int p = blockIdx.x*64 + j*4 + sub;       // coalesced across the 256 threads
    if (p < HW){ float v = base[(size_t)p*CH + ch]; s += v; q += v*v; }
  }
  atomicAdd(&ls[ch], s); atomicAdd(&lq[ch], q);
  __syncthreads();
  if (tid < CH){
    int g = bn_group(img);
    atomicAdd(&stats[(g*CH + tid)*2 + 0], ls[tid]);
    atomicAdd(&stats[(g*CH + tid)*2 + 1], lq[tid]);
  }
}

// BN + LeakyReLU (+ optional 2x2 maxpool), f32 NHWC -> f16 NHWC
__global__ void apply_bn(const float* __restrict__ buf, const float* __restrict__ ss,
                         _Float16* __restrict__ act, int Hi, int Wi, int dopool){
  int Ho = dopool ? Hi/2 : Hi, Wo = dopool ? Wi/2 : Wi;
  int idx = blockIdx.x*256 + threadIdx.x;
  if (idx >= NIMG*Ho*Wo*CH) return;
  int o = idx & 63; int t = idx >> 6;
  int ox = t % Wo; t /= Wo;
  int oy = t % Ho; int img = t / Ho;
  int g = bn_group(img);
  float sc = ss[(g*CH + o)*2 + 0], sh = ss[(g*CH + o)*2 + 1];
  const float* base = buf + (size_t)img*Hi*Wi*CH;
  float m = NEG_BIG;
  int n = dopool ? 2 : 1;
  for (int py = 0; py < n; ++py)
    for (int px = 0; px < n; ++px){
      int yy = dopool ? 2*oy + py : oy;
      int xx = dopool ? 2*ox + px : ox;
      float v = base[(size_t)(yy*Wi + xx)*CH + o] * sc + sh;
      v = (v >= 0.f) ? v : 0.2f * v;
      m = fmaxf(m, v);
    }
  act[idx] = (_Float16)m;
}

// per-(img,pixel) L2-normalize over 64 channels (in place) -> cosine-ready features
__global__ void normalize_feat(_Float16* feat){
  int idx = blockIdx.x*256 + threadIdx.x;
  if (idx >= NIMG*HW2) return;
  _Float16* p = feat + (size_t)idx*CH;
  float v[CH], ssq = 0.f;
  #pragma unroll
  for (int i = 0; i < CH; ++i){ v[i] = (float)p[i]; ssq += v[i]*v[i]; }
  float rs = rsqrtf(ssq + 1e-24f);
  #pragma unroll
  for (int i = 0; i < CH; ++i) p[i] = (_Float16)(v[i]*rs);
}

// ----------------- cosine-sim GEMM (WMMA, K=64) + running per-query top-3, sum -> out
// one wave per (b, nc, 16-query tile); D[m(16) x q(16)] = Shat * Qhat^T
__global__ void __launch_bounds__(256) sim_topk(const _Float16* __restrict__ feat,
                                                float* __restrict__ out){
  int lane = threadIdx.x & 31;
  int gw   = blockIdx.x*8 + (threadIdx.x >> 5);
  if (gw >= NQIMG*NCLS*NQT) return;
  int qt = gw % NQT; int t = gw / NQT;
  int nc = t % NCLS; int b = t / NCLS;

  int q = qt*16 + (lane & 15);
  bool qv = q < HW2;
  const _Float16* qbase = feat + ((size_t)b*HW2 + (qv ? q : 0))*CH;
  int cib = (lane & 16) ? 16 : 0;
  v16h bf0 = qv ? *(const v16h*)(qbase + cib)      : zero16();   // hoisted B-frags
  v16h bf1 = qv ? *(const v16h*)(qbase + 32 + cib) : zero16();

  const _Float16* sbase = feat + (size_t)(NQIMG + nc*SHOT)*HW2*CH;
  int arow = lane & 15;
  int aoff = (lane & 16) ? 16 : 0;                // A-frag K sub-range per lane half
  float t0 = NEG_BIG, t1 = NEG_BIG, t2 = NEG_BIG;

  for (int mt = 0; mt < NMT; ++mt){
    int m = mt*16 + arow;
    if (m + 16 < M_SUP)                            // prefetch next tile's A rows
      __builtin_prefetch(sbase + (size_t)(m + 16)*CH, 0, 1);
    v16h a0, a1;
    if (m < M_SUP){
      const _Float16* ap = sbase + (size_t)m*CH;
      v8h l0 = *(const v8h*)(ap + aoff);          // K 0..7 / 16..23
      v8h h0 = *(const v8h*)(ap + aoff + 8);      // K 8..15 / 24..31
      v8h l1 = *(const v8h*)(ap + 32 + aoff);
      v8h h1 = *(const v8h*)(ap + 32 + aoff + 8);
      #pragma unroll
      for (int i = 0; i < 8; ++i){ a0[i] = l0[i]; a0[8+i] = h0[i]; a1[i] = l1[i]; a1[8+i] = h1[i]; }
    } else { a0 = zero16(); a1 = zero16(); }
    v8f c = {};
    c = wmma_f16(a0, bf0, c);
    c = wmma_f16(a1, bf1, c);
    int mbase = mt*16 + ((lane & 16) ? 8 : 0);    // D rows r / r+8 per lane half
    #pragma unroll
    for (int r = 0; r < 8; ++r){
      // tail-mask via select (v_cndmask), then branchless min/max insert
      float v = (mbase + r < M_SUP) ? c[r] : NEG_BIG;
      top3_insert(v, t0, t1, t2);
    }
  }
  // merge with partner lane (same q, other half of m rows): branchless
  float p0 = __shfl_xor(t0, 16, 32), p1 = __shfl_xor(t1, 16, 32), p2 = __shfl_xor(t2, 16, 32);
  top3_insert(p0, t0, t1, t2); top3_insert(p1, t0, t1, t2); top3_insert(p2, t0, t1, t2);
  float sum = qv ? (t0 + t1 + t2) : 0.f;
  sum += __shfl_xor(sum, 1, 32); sum += __shfl_xor(sum, 2, 32);
  sum += __shfl_xor(sum, 4, 32); sum += __shfl_xor(sum, 8, 32);
  if (lane == 0) atomicAdd(&out[b*NCLS + nc], sum);
}

// ================================================================= host launch
extern "C" void kernel_launch(void* const* d_in, const int* in_sizes, int n_in,
                              void* d_out, int out_size, void* d_ws, size_t ws_size,
                              hipStream_t stream){
  (void)in_sizes; (void)n_in; (void)out_size; (void)ws_size;
  const float* input1 = (const float*)d_in[0];
  const float* input2 = (const float*)d_in[1];
  const float* w1 = (const float*)d_in[2];
  const float* g1 = (const float*)d_in[3];
  const float* b1 = (const float*)d_in[4];
  const float* w2 = (const float*)d_in[5];
  const float* g2 = (const float*)d_in[6];
  const float* b2 = (const float*)d_in[7];
  const float* w3 = (const float*)d_in[8];
  const float* g3 = (const float*)d_in[9];
  const float* b3 = (const float*)d_in[10];
  const float* w4 = (const float*)d_in[11];
  const float* g4 = (const float*)d_in[12];
  const float* b4 = (const float*)d_in[13];
  float* out = (float*)d_out;

  char* ws = (char*)d_ws;
  size_t off = 0;
  auto alloc = [&](size_t bytes) -> char* {
    char* p = ws + off; off += (bytes + 255) & ~(size_t)255; return p;
  };
  float*    stats   = (float*)   alloc((size_t)4*6*CH*2*sizeof(float));
  float*    ssb     = (float*)   alloc((size_t)4*6*CH*2*sizeof(float));
  _Float16* wp2     = (_Float16*)alloc((size_t)WPK_BYTES);
  _Float16* wp3     = (_Float16*)alloc((size_t)WPK_BYTES);
  _Float16* wp4     = (_Float16*)alloc((size_t)WPK_BYTES);
  _Float16* act1    = (_Float16*)alloc((size_t)NIMG*HW1*CH*2);
  float*    convbuf = (float*)   alloc((size_t)NIMG*HW1*CH*4);
  _Float16* act2    = (_Float16*)alloc((size_t)NIMG*HW2*CH*2);
  _Float16* act3    = (_Float16*)alloc((size_t)NIMG*HW2*CH*2);
  _Float16* act4    = (_Float16*)alloc((size_t)NIMG*HW2*CH*2);

  const int STRIDE = 6*CH*2;   // floats per layer in stats/ssb

  zero_init<<<12, 256, 0, stream>>>(stats, out);
  pack_w<<<(WPK_ELEMS+255)/256, 256, 0, stream>>>(w2, wp2);
  pack_w<<<(WPK_ELEMS+255)/256, 256, 0, stream>>>(w3, wp3);
  pack_w<<<(WPK_ELEMS+255)/256, 256, 0, stream>>>(w4, wp4);

  // layer 1: direct conv, fused two-pass BN, pool -> act1 (f16 NHWC, 42x42)
  conv1_stats<<<dim3((HW0+63)/64, NIMG), 256, 0, stream>>>(input1, input2, w1, stats);
  bn_finalize<<<6, 64, 0, stream>>>(stats, ssb, g1, b1, HW0);
  apply1<<<(NIMG*HW1*CH+255)/256, 256, 0, stream>>>(input1, input2, w1, ssb, act1);

  // layer 2: WMMA conv 42x42 -> BN -> pool -> act2 (21x21)
  conv_wmma<H1, NT1><<<dim3((NT1+7)/8, NIMG), 256, WPK_BYTES, stream>>>(act1, wp2, convbuf);
  stats_buf<<<dim3((HW1+63)/64, NIMG), 256, 0, stream>>>(convbuf, stats + STRIDE, HW1);
  bn_finalize<<<6, 64, 0, stream>>>(stats + STRIDE, ssb + STRIDE, g2, b2, HW1);
  apply_bn<<<(NIMG*HW2*CH+255)/256, 256, 0, stream>>>(convbuf, ssb + STRIDE, act2, H1, H1, 1);

  // layer 3: WMMA conv 21x21 -> BN -> act3
  conv_wmma<H2, NT2><<<dim3((NT2+7)/8, NIMG), 256, WPK_BYTES, stream>>>(act2, wp3, convbuf);
  stats_buf<<<dim3((HW2+63)/64, NIMG), 256, 0, stream>>>(convbuf, stats + 2*STRIDE, HW2);
  bn_finalize<<<6, 64, 0, stream>>>(stats + 2*STRIDE, ssb + 2*STRIDE, g3, b3, HW2);
  apply_bn<<<(NIMG*HW2*CH+255)/256, 256, 0, stream>>>(convbuf, ssb + 2*STRIDE, act3, H2, H2, 0);

  // layer 4: WMMA conv 21x21 -> BN -> act4
  conv_wmma<H2, NT2><<<dim3((NT2+7)/8, NIMG), 256, WPK_BYTES, stream>>>(act3, wp4, convbuf);
  stats_buf<<<dim3((HW2+63)/64, NIMG), 256, 0, stream>>>(convbuf, stats + 3*STRIDE, HW2);
  bn_finalize<<<6, 64, 0, stream>>>(stats + 3*STRIDE, ssb + 3*STRIDE, g4, b4, HW2);
  apply_bn<<<(NIMG*HW2*CH+255)/256, 256, 0, stream>>>(convbuf, ssb + 3*STRIDE, act4, H2, H2, 0);

  // cosine similarity + top-3
  normalize_feat<<<(NIMG*HW2+255)/256, 256, 0, stream>>>(act4);
  sim_topk<<<(NQIMG*NCLS*NQT+7)/8, 256, 0, stream>>>(act4, out);
}
